// Hierarchical12GNN_37177236914922
// MI455X (gfx1250) — matrix operand order repaired and checked by reading, not compiled, not run on hardware
//
#include <hip/hip_runtime.h>
#include <hip/hip_bf16.h>

// ---------------------------------------------------------------------------
// CDNA5 (gfx1250) WMMA implementation of the hierarchical 1-2-GNN.
// All dense matmuls use v_wmma_f32_16x16x32_bf16 (wave32, one wave per 16x16
// output tile).  Activations/weights are kept bf16-resident in layouts that
// make every WMMA fragment load a contiguous b128:
//   A 16x32 tile : lane(r,hi) rows, K run {k0+16hi+0..15}
//   B 32x16 tile : from transposed weights Wt[n][k], K run {k0+16hi+0..15}
// Sparse aggregation uses linearity:  sum(h[src] @ W2) == (sum h[src]) @ W2,
// so the graph part is only fp32 row scatter-adds + a bf16 convert pass.
// ---------------------------------------------------------------------------

typedef __bf16 bf16;
typedef __attribute__((ext_vector_type(16))) bf16  v16bf;
typedef __attribute__((ext_vector_type(8)))  bf16  v8bf;
typedef __attribute__((ext_vector_type(8)))  float v8f;

#define NNODE 4096      // B * N_PER
#define NPAIR 129024    // B * P_PER
#define HID   128
#define INDIM 64
#define NB    64
#define NPER  64
#define PPER  2016
#define OUTD  10

// ---- fragment loaders ------------------------------------------------------
// A fragment: 16x32 bf16 tile, this lane's row at `row` (bf16*, stride = K)
__device__ __forceinline__ v16bf load_a(const bf16* __restrict__ row, int k0, int hi) {
    return *(const v16bf*)(row + k0 + hi * 16);
}
// B fragment: 32x16 tile from transposed weights Wt[N][K]; `col` = Wt row ptr
__device__ __forceinline__ v16bf load_b(const bf16* __restrict__ col, int k0, int hi) {
    return *(const v16bf*)(col + k0 + hi * 16);
}

// ---- small helpers ---------------------------------------------------------
__global__ void zero_f(float* __restrict__ p, long n) {
    long t = (long)blockIdx.x * blockDim.x + threadIdx.x;
    if (t < n) p[t] = 0.0f;
}

__global__ void cvt_bf(const float* __restrict__ x, bf16* __restrict__ y, long n) {
    long t = (long)blockIdx.x * blockDim.x + threadIdx.x;
    if (t < n) y[t] = (bf16)x[t];
}

// transpose + convert:  Wt[n*K + k] = (bf16) W[k*N + n]
__global__ void wt_prep(const float* __restrict__ W, bf16* __restrict__ Wt, int K, int N) {
    int t = blockIdx.x * blockDim.x + threadIdx.x;
    if (t >= K * N) return;
    int k = t / N, n = t % N;
    Wt[(long)n * K + k] = (bf16)W[(long)k * N + n];
}

__global__ void add_relu_dual(const float* __restrict__ a, const float* __restrict__ b,
                              float* __restrict__ o, bf16* __restrict__ ob, long n) {
    long t = (long)blockIdx.x * blockDim.x + threadIdx.x;
    if (t < n) {
        float v = a[t] + b[t];
        v = v > 0.0f ? v : 0.0f;
        o[t] = v;
        ob[t] = (bf16)v;
    }
}

// S[dst[e], :] += X[src[e], :]   (one thread per (edge, col)), fp32 atomics
__global__ void scatter_rows(const float* __restrict__ X, const int* __restrict__ src,
                             const int* __restrict__ dst, long nE, int dim,
                             float* __restrict__ S) {
    long t = (long)blockIdx.x * blockDim.x + threadIdx.x;
    long total = nE * (long)dim;
    if (t >= total) return;
    long e = t / dim;
    int  c = (int)(t - e * dim);
    atomicAdd(&S[(long)dst[e] * dim + c], X[(long)src[e] * dim + c]);
}

// ---------------------------------------------------------------------------
// out = act( H @ W1 + S @ W2 ),  H,S bf16 [M,K], W*t bf16 transposed [128,K],
// out fp32 [M,128] (+ optional bf16 copy).  grid = (M/16)*8, block = 32.
// ---------------------------------------------------------------------------
template <int K>
__global__ __launch_bounds__(32)
void mp_gemm(const bf16* __restrict__ Hb, const bf16* __restrict__ Sb,
             const bf16* __restrict__ W1t, const bf16* __restrict__ W2t,
             float* __restrict__ out, bf16* __restrict__ obf, int do_relu) {
    int tm = blockIdx.x >> 3, tn = blockIdx.x & 7;
    int lane = threadIdx.x & 31;
    int r = lane & 15, hi = lane >> 4;
    const bf16* hrow = Hb + (long)(tm * 16 + r) * K;
    const bf16* srow = Sb + (long)(tm * 16 + r) * K;
    const bf16* w1c  = W1t + (long)(tn * 16 + r) * K;
    const bf16* w2c  = W2t + (long)(tn * 16 + r) * K;

    v8f acc = {};
#pragma unroll
    for (int k0 = 0; k0 < K; k0 += 32)
        acc = __builtin_amdgcn_wmma_f32_16x16x32_bf16(false, load_a(hrow, k0, hi),
                                                      false, load_b(w1c, k0, hi),
                                                      (short)0, acc, false, false);
#pragma unroll
    for (int k0 = 0; k0 < K; k0 += 32)
        acc = __builtin_amdgcn_wmma_f32_16x16x32_bf16(false, load_a(srow, k0, hi),
                                                      false, load_b(w2c, k0, hi),
                                                      (short)0, acc, false, false);
#pragma unroll
    for (int i = 0; i < 8; ++i) {
        float v = acc[i];
        if (do_relu && v < 0.0f) v = 0.0f;
        long idx = (long)(tm * 16 + hi * 8 + i) * HID + tn * 16 + r;
        out[idx] = v;
        if (obf) obf[idx] = (bf16)v;
    }
}

// ---------------------------------------------------------------------------
// First 2-set layer, fused:  A_row(p) = [ h[pu[p]] , h[pv[p]] ]  (K = 256)
//   Z[p]  = A_row @ W1[0:256] + iso[p] * W1[256]   (pre-activation -> P0)
//   Mm[p] = A_row @ W2[0:256] + iso[p] * W2[256]   (message -> P1)
// hb: bf16 [4096,128]; W*t: bf16 [128,256]; W*f: original fp32 [257,128].
// ---------------------------------------------------------------------------
__global__ __launch_bounds__(32)
void pair_gemm(const bf16* __restrict__ hb,
               const bf16* __restrict__ W1t, const bf16* __restrict__ W2t,
               const float* __restrict__ W1f, const float* __restrict__ W2f,
               const float* __restrict__ iso,
               const int* __restrict__ pu, const int* __restrict__ pv,
               float* __restrict__ Z, float* __restrict__ Mm) {
    int tm = blockIdx.x >> 3, tn = blockIdx.x & 7;
    int lane = threadIdx.x & 31;
    int r = lane & 15, hi = lane >> 4;
    const bf16* urow = hb + (long)pu[tm * 16 + r] * HID;
    const bf16* vrow = hb + (long)pv[tm * 16 + r] * HID;
    const bf16* w1c  = W1t + (long)(tn * 16 + r) * 256;
    const bf16* w2c  = W2t + (long)(tn * 16 + r) * 256;

    v8f acc1 = {}, acc2 = {};
#pragma unroll
    for (int k0 = 0; k0 < 256; k0 += 32) {
        const bf16* arow = (k0 < HID) ? urow : (vrow - HID);
        v16bf a = load_a(arow, k0, hi);
        acc1 = __builtin_amdgcn_wmma_f32_16x16x32_bf16(false, a, false, load_b(w1c, k0, hi),
                                                       (short)0, acc1, false, false);
        acc2 = __builtin_amdgcn_wmma_f32_16x16x32_bf16(false, a, false, load_b(w2c, k0, hi),
                                                       (short)0, acc2, false, false);
    }
    int col = tn * 16 + r;
    float i1 = W1f[256 * HID + col];
    float i2 = W2f[256 * HID + col];
#pragma unroll
    for (int i = 0; i < 8; ++i) {
        int prow = tm * 16 + hi * 8 + i;
        float is = iso[prow];
        Z[(long)prow * HID + col]  = acc1[i] + is * i1;
        Mm[(long)prow * HID + col] = acc2[i] + is * i2;
    }
}

// segment sum over contiguous equal-size segments
__global__ void seg_sum(const float* __restrict__ X, float* __restrict__ out,
                        int rows, int dim, int nseg, int out_stride, int out_off) {
    int t = blockIdx.x * blockDim.x + threadIdx.x;
    if (t >= nseg * dim) return;
    int b = t / dim, c = t % dim;
    const float* p = X + (long)b * rows * dim + c;
    float s = 0.0f;
    for (int i = 0; i < rows; ++i) s += p[(long)i * dim];
    out[(long)b * out_stride + out_off + c] = s;
}

// small dense classifier layers (tiny FLOPs -> plain VALU is fine)
__global__ void fc(const float* __restrict__ X, const float* __restrict__ W,
                   const float* __restrict__ bias, float* __restrict__ out,
                   int M, int N, int K, int do_relu) {
    int t = blockIdx.x * blockDim.x + threadIdx.x;
    if (t >= M * N) return;
    int m = t / N, n = t % N;
    float s = bias[n];
    for (int k = 0; k < K; ++k) s += X[(long)m * K + k] * W[(long)k * N + n];
    if (do_relu && s < 0.0f) s = 0.0f;
    out[t] = s;
}

// ---------------------------------------------------------------------------
extern "C" void kernel_launch(void* const* d_in, const int* in_sizes, int n_in,
                              void* d_out, int out_size, void* d_ws, size_t ws_size,
                              hipStream_t stream) {
    (void)n_in; (void)out_size; (void)ws_size;
    const float* x      = (const float*)d_in[0];
    const float* g1W1_0 = (const float*)d_in[1];
    const float* g1W2_0 = (const float*)d_in[2];
    const float* g1W1_1 = (const float*)d_in[3];
    const float* g1W2_1 = (const float*)d_in[4];
    const float* g1W1_2 = (const float*)d_in[5];
    const float* g1W2_2 = (const float*)d_in[6];
    const float* kW1_0  = (const float*)d_in[7];
    const float* kW2_0  = (const float*)d_in[8];
    const float* kW1_1  = (const float*)d_in[9];
    const float* kW2_1  = (const float*)d_in[10];
    const float* cW0    = (const float*)d_in[11];
    const float* cb0    = (const float*)d_in[12];
    const float* cW1    = (const float*)d_in[13];
    const float* cb1    = (const float*)d_in[14];
    const float* iso    = (const float*)d_in[15];
    const int*   ei     = (const int*)d_in[16];
    const int*   pu     = (const int*)d_in[18];
    const int*   pv     = (const int*)d_in[19];
    const int*   te     = (const int*)d_in[20];
    float*       out    = (float*)d_out;

    long E  = in_sizes[16] / 2;   // directed graph edges
    long E2 = in_sizes[20] / 2;   // 2-set edges

    // ---- workspace layout ----
    char* base = (char*)d_ws;
    auto alloc_f = [&](long n) { float* p = (float*)base; base += n * 4; return p; };
    auto alloc_b = [&](long n) { bf16*  p = (bf16*) base; base += n * 2; return p; };

    float* P0  = alloc_f((long)NPAIR * HID);
    float* P1  = alloc_f((long)NPAIR * HID);
    float* P2  = alloc_f((long)NPAIR * HID);
    float* hAf = alloc_f((long)NNODE * HID);
    float* hBf = alloc_f((long)NNODE * HID);
    float* Sf  = alloc_f((long)NNODE * HID);
    float* emb = alloc_f(NB * 2 * HID);
    float* hc  = alloc_f(NB * HID);
    bf16*  H2b = alloc_b((long)NPAIR * HID);
    bf16*  S2b = alloc_b((long)NPAIR * HID);
    bf16*  hAb = alloc_b((long)NNODE * HID);
    bf16*  hBb = alloc_b((long)NNODE * HID);
    bf16*  Sb  = alloc_b((long)NNODE * HID);
    bf16*  xb  = alloc_b((long)NNODE * INDIM);
    bf16*  Wt10 = alloc_b(INDIM * HID), *Wt20 = alloc_b(INDIM * HID);
    bf16*  Wt11 = alloc_b(HID * HID),   *Wt21 = alloc_b(HID * HID);
    bf16*  Wt12 = alloc_b(HID * HID),   *Wt22 = alloc_b(HID * HID);
    bf16*  Kt10 = alloc_b(256 * HID),   *Kt20 = alloc_b(256 * HID);
    bf16*  Kt11 = alloc_b(HID * HID),   *Kt21 = alloc_b(HID * HID);

    const int T = 256;
    auto cdiv = [](long a, long b) { return (int)((a + b - 1) / b); };

    // ---- one-time prep: bf16 transposed weights + bf16 input features ----
    wt_prep<<<cdiv(INDIM * HID, T), T, 0, stream>>>(g1W1_0, Wt10, INDIM, HID);
    wt_prep<<<cdiv(INDIM * HID, T), T, 0, stream>>>(g1W2_0, Wt20, INDIM, HID);
    wt_prep<<<cdiv(HID * HID, T), T, 0, stream>>>(g1W1_1, Wt11, HID, HID);
    wt_prep<<<cdiv(HID * HID, T), T, 0, stream>>>(g1W2_1, Wt21, HID, HID);
    wt_prep<<<cdiv(HID * HID, T), T, 0, stream>>>(g1W1_2, Wt12, HID, HID);
    wt_prep<<<cdiv(HID * HID, T), T, 0, stream>>>(g1W2_2, Wt22, HID, HID);
    wt_prep<<<cdiv(256 * HID, T), T, 0, stream>>>(kW1_0, Kt10, 256, HID);  // rows 0..255
    wt_prep<<<cdiv(256 * HID, T), T, 0, stream>>>(kW2_0, Kt20, 256, HID);
    wt_prep<<<cdiv(HID * HID, T), T, 0, stream>>>(kW1_1, Kt11, HID, HID);
    wt_prep<<<cdiv(HID * HID, T), T, 0, stream>>>(kW2_1, Kt21, HID, HID);
    cvt_bf<<<cdiv((long)NNODE * INDIM, T), T, 0, stream>>>(x, xb, (long)NNODE * INDIM);

    // ---- graph MP layer 0 (K = 64) ----
    zero_f<<<cdiv((long)NNODE * INDIM, T), T, 0, stream>>>(Sf, (long)NNODE * INDIM);
    scatter_rows<<<cdiv(E * INDIM, T), T, 0, stream>>>(x, ei, ei + E, E, INDIM, Sf);
    cvt_bf<<<cdiv((long)NNODE * INDIM, T), T, 0, stream>>>(Sf, Sb, (long)NNODE * INDIM);
    mp_gemm<INDIM><<<(NNODE / 16) * 8, 32, 0, stream>>>(xb, Sb, Wt10, Wt20, hAf, hAb, 1);
    // ---- graph MP layer 1 ----
    zero_f<<<cdiv((long)NNODE * HID, T), T, 0, stream>>>(Sf, (long)NNODE * HID);
    scatter_rows<<<cdiv(E * HID, T), T, 0, stream>>>(hAf, ei, ei + E, E, HID, Sf);
    cvt_bf<<<cdiv((long)NNODE * HID, T), T, 0, stream>>>(Sf, Sb, (long)NNODE * HID);
    mp_gemm<HID><<<(NNODE / 16) * 8, 32, 0, stream>>>(hAb, Sb, Wt11, Wt21, hBf, hBb, 1);
    // ---- graph MP layer 2 ----
    zero_f<<<cdiv((long)NNODE * HID, T), T, 0, stream>>>(Sf, (long)NNODE * HID);
    scatter_rows<<<cdiv(E * HID, T), T, 0, stream>>>(hBf, ei, ei + E, E, HID, Sf);
    cvt_bf<<<cdiv((long)NNODE * HID, T), T, 0, stream>>>(Sf, Sb, (long)NNODE * HID);
    mp_gemm<HID><<<(NNODE / 16) * 8, 32, 0, stream>>>(hBb, Sb, Wt12, Wt22, hAf, hAb, 1);
    // ---- graph pooling -> emb[:, 0:128] ----
    seg_sum<<<cdiv(NB * HID, T), T, 0, stream>>>(hAf, emb, NPER, HID, NB, 2 * HID, 0);

    // ---- 2-set layer 0: Z -> P0, message -> P1 ----
    pair_gemm<<<(NPAIR / 16) * 8, 32, 0, stream>>>(hAb, Kt10, Kt20, kW1_0, kW2_0,
                                                   iso, pu, pv, P0, P1);
    zero_f<<<cdiv((long)NPAIR * HID, T), T, 0, stream>>>(P2, (long)NPAIR * HID);
    scatter_rows<<<cdiv(E2 * HID, T), T, 0, stream>>>(P1, te, te + E2, E2, HID, P2);
    add_relu_dual<<<cdiv((long)NPAIR * HID, T), T, 0, stream>>>(P0, P2, P0, H2b,
                                                                (long)NPAIR * HID);
    // ---- 2-set layer 1 ----
    zero_f<<<cdiv((long)NPAIR * HID, T), T, 0, stream>>>(P1, (long)NPAIR * HID);
    scatter_rows<<<cdiv(E2 * HID, T), T, 0, stream>>>(P0, te, te + E2, E2, HID, P1);
    cvt_bf<<<cdiv((long)NPAIR * HID, T), T, 0, stream>>>(P1, S2b, (long)NPAIR * HID);
    mp_gemm<HID><<<(NPAIR / 16) * 8, 32, 0, stream>>>(H2b, S2b, Kt11, Kt21, P2,
                                                      (bf16*)nullptr, 1);
    // ---- 2-set pooling -> emb[:, 128:256] ----
    seg_sum<<<cdiv(NB * HID, T), T, 0, stream>>>(P2, emb, PPER, HID, NB, 2 * HID, HID);

    // ---- classifier ----
    fc<<<cdiv(NB * HID, T), T, 0, stream>>>(emb, cW0, cb0, hc, NB, HID, 2 * HID, 1);
    fc<<<cdiv(NB * OUTD, T), T, 0, stream>>>(hc, cW1, cb1, out, NB, OUTD, HID, 0);
}
